// Attention_18511309045886
// MI455X (gfx1250) — matrix operand order repaired and checked by reading, not compiled, not run on hardware
//
#include <hip/hip_runtime.h>
#include <hip/hip_bf16.h>

// Problem constants (B=2, N=4096, C=512, H=8, D=64)
#define BB 2
#define NN 4096
#define CC 512
#define HH 8
#define DD 64
#define SCALE 0.125f   // 64^-0.5, folded into Q at QKV time

typedef __attribute__((ext_vector_type(16))) __bf16 v16bf;
typedef __attribute__((ext_vector_type(8)))  __bf16 v8bf;
typedef __attribute__((ext_vector_type(8)))  float  v8f;

__device__ __forceinline__ __bf16 to_bf16(float f) {
  unsigned u = __float_as_uint(f);
  u += 0x7FFFu + ((u >> 16) & 1u);           // round-to-nearest-even
  return __builtin_bit_cast(__bf16, (unsigned short)(u >> 16));
}
// cheap truncating convert: used only for P in [0,1] (err <= 2^-8)
__device__ __forceinline__ __bf16 to_bf16_rz(float f) {
  return __builtin_bit_cast(__bf16, (unsigned short)(__float_as_uint(f) >> 16));
}

__device__ __forceinline__ v8f wmma_bf16(v16bf a, v16bf b, v8f c) {
  // D = A(16x32 bf16) x B(32x16 bf16) + C(16x16 f32)
  return __builtin_amdgcn_wmma_f32_16x16x32_bf16(false, a, false, b, (short)0, c, false, false);
}

// A-fragment (16x32, 16-bit): lane = hg*16+m holds row m; halfwords 0..7 are
// K = klo..klo+7, halfwords 8..15 are K = khi..khi+7 (klo = kb + hg*16, khi = kb+hg*16+8)
__device__ __forceinline__ v16bf load_a_frag(const __bf16* row, int klo, int khi) {
  v8bf lo = *(const v8bf*)(row + klo);
  v8bf hi = *(const v8bf*)(row + khi);
  v16bf a;
#pragma unroll
  for (int i = 0; i < 8; ++i) { a[i] = lo[i]; a[i + 8] = hi[i]; }
  return a;
}

__device__ __forceinline__ float hmax16(float v) {
  v = fmaxf(v, __shfl_xor(v, 1));
  v = fmaxf(v, __shfl_xor(v, 2));
  v = fmaxf(v, __shfl_xor(v, 4));
  v = fmaxf(v, __shfl_xor(v, 8));
  return v;
}
__device__ __forceinline__ float hsum16(float v) {
  v += __shfl_xor(v, 1);
  v += __shfl_xor(v, 2);
  v += __shfl_xor(v, 4);
  v += __shfl_xor(v, 8);
  return v;
}

// ---------------------------------------------------------------- cvt f32->bf16
__global__ __launch_bounds__(256) void cvt_bf16_kernel(const float* __restrict__ src,
                                                       __bf16* __restrict__ dst, int n) {
  int i = blockIdx.x * blockDim.x + threadIdx.x;
  int stride = gridDim.x * blockDim.x;
  for (; i < n; i += stride) dst[i] = to_bf16(src[i]);
}

// ------------------------------------------------- QKV GEMM + bias + RoPE + scatter
// Per wave: 16x64 tile of (8192 x 1536) output. A = x_bf (row m, K=512),
// B[k][n] = Wqkv_bf[n][k] (contiguous 32B per lane). Q is pre-scaled by 1/sqrt(d).
__global__ __launch_bounds__(256) void qkv_rope_kernel(const __bf16* __restrict__ xb,
                                                       const __bf16* __restrict__ Wb,
                                                       const float* __restrict__ bqkv,
                                                       __bf16* __restrict__ Qb,
                                                       __bf16* __restrict__ Kb,
                                                       __bf16* __restrict__ Vt) {
  const int wave = threadIdx.x >> 5, lane = threadIdx.x & 31;
  const int hg = lane >> 4, ln = lane & 15;
  const int wid = blockIdx.x * 8 + wave;          // 12288 wave-tiles
  const int m0 = (wid % 512) * 16;                // row tile
  const int n0 = (wid / 512) * 64;                // col tile (64 = one head span)

  v8f acc[4];
#pragma unroll
  for (int f = 0; f < 4; ++f)
#pragma unroll
    for (int r = 0; r < 8; ++r) acc[f][r] = 0.0f;

  const __bf16* arow = xb + (size_t)(m0 + ln) * CC;
  for (int kb = 0; kb < CC; kb += 32) {
    v16bf a = load_a_frag(arow, kb + hg * 16, kb + hg * 16 + 8);
#pragma unroll
    for (int f = 0; f < 4; ++f) {
      int col = n0 + 16 * f + ln;
      v16bf b = *(const v16bf*)(Wb + (size_t)col * CC + kb + hg * 16);
      acc[f] = wmma_bf16(a, b, acc[f]);
    }
  }

  const int which = n0 / CC;             // 0=q 1=k 2=v
  const int h = (n0 % CC) / DD;
  const int bI = m0 / NN;
  const int pos0 = m0 % NN;

#pragma unroll
  for (int f = 0; f < 4; ++f) {
    const int col = n0 + 16 * f + ln;
    const int d = 16 * f + ln;                 // d within head, 0..63
    const float bias = bqkv[col];
    const bool even = ((d & 1) == 0);
    const int p2 = d >> 1;
    const float freq = __expf((float)(-2 * p2) * 0.14391565f);  // ln(1e4)/64
#pragma unroll
    for (int r = 0; r < 8; ++r) {
      float v = acc[f][r] + bias;
      float pv = __shfl_xor(v, 1);             // partner column (d ^ 1)
      const int pos = pos0 + r + 8 * hg;
      if (which < 2) {                          // RoPE on q,k
        float sn, cs;
        __sincosf((float)pos * freq, &sn, &cs);
        float nv = even ? (v * cs - pv * sn) : (v * cs + pv * sn);
        if (which == 0) nv *= SCALE;            // fold softmax scale into Q
        __bf16* dst = (which == 0) ? Qb : Kb;
        dst[((size_t)(bI * HH + h) * NN + pos) * DD + d] = to_bf16(nv);
      } else {                                  // V stored transposed (b,h,d,pos)
        Vt[((size_t)(bI * HH + h) * DD + d) * NN + pos] = to_bf16(v);
      }
    }
  }
}

// ------------------------------------------------- flash attention, 1 wave / 16 q-rows
// 64-key tiles: 16 WMMAs per iteration, softmax reductions/rescale amortized 2x.
__global__ __launch_bounds__(256) void attn_kernel(const __bf16* __restrict__ Qb,
                                                   const __bf16* __restrict__ Kb,
                                                   const __bf16* __restrict__ Vt,
                                                   __bf16* __restrict__ Ob) {
  __shared__ __bf16 plds[8][16 * 64];           // per-wave P transpose buffer (16 KB)
  const int wave = threadIdx.x >> 5, lane = threadIdx.x & 31;
  const int hg = lane >> 4, ln = lane & 15;
  const int wid = blockIdx.x * 8 + wave;        // 4096 wave-tiles
  const int bh = wid >> 8;                      // 0..15 (b*8+h)
  const int q0 = (wid & 255) * 16;

  const __bf16* Qp = Qb + (size_t)bh * NN * DD;
  const __bf16* Kp = Kb + (size_t)bh * NN * DD;
  const __bf16* Vp = Vt + (size_t)bh * DD * NN;

  // Q fragments (row = q0+ln, d-halves 0..31 / 32..63), loaded once
  const __bf16* qrow = Qp + (size_t)(q0 + ln) * DD;
  v16bf qa0 = load_a_frag(qrow, hg * 16, hg * 16 + 8);
  v16bf qa1 = load_a_frag(qrow, 32 + hg * 16, 40 + hg * 16);

  float mst[8], lst[8];
  v8f o[4];
#pragma unroll
  for (int r = 0; r < 8; ++r) { mst[r] = -1e30f; lst[r] = 0.0f; }
#pragma unroll
  for (int f = 0; f < 4; ++f)
#pragma unroll
    for (int r = 0; r < 8; ++r) o[f][r] = 0.0f;

  __bf16* myP = plds[wave];

  for (int j0 = 0; j0 < NN; j0 += 64) {
    // prefetch next tile's K rows / V rows while we crunch this one
    if (j0 + 64 < NN) {
      __builtin_prefetch(Kp + (size_t)(j0 + 64 + ln + 16 * hg) * DD, 0, 3);
      __builtin_prefetch(Vp + (size_t)(ln + 16 * hg) * NN + j0 + 64, 0, 3);
    }

    // S = Q(16x64) . K_tile^T(64x64)  -> four 16x16 f32 fragments
    v8f s[4];
#pragma unroll
    for (int g = 0; g < 4; ++g) {
#pragma unroll
      for (int r = 0; r < 8; ++r) s[g][r] = 0.0f;
      const __bf16* krow = Kp + (size_t)(j0 + 16 * g + ln) * DD;
      v16bf kb0 = *(const v16bf*)(krow + hg * 16);
      s[g] = wmma_bf16(qa0, kb0, s[g]);
      v16bf kb1 = *(const v16bf*)(krow + 32 + hg * 16);
      s[g] = wmma_bf16(qa1, kb1, s[g]);
    }

    // online softmax (row = r + 8*hg, fragment g covers keys j0+16g+ln)
    float corr[8];
#pragma unroll
    for (int r = 0; r < 8; ++r) {
      float mx = fmaxf(fmaxf(s[0][r], s[1][r]), fmaxf(s[2][r], s[3][r]));
      mx = hmax16(mx);
      float mnew = fmaxf(mst[r], mx);
      float p0 = __expf(s[0][r] - mnew);
      float p1 = __expf(s[1][r] - mnew);
      float p2 = __expf(s[2][r] - mnew);
      float p3 = __expf(s[3][r] - mnew);
      float rs = hsum16(((p0 + p1) + (p2 + p3)));
      float cr = __expf(mst[r] - mnew);
      lst[r] = lst[r] * cr + rs;
      mst[r] = mnew;
      corr[r] = cr;
      const int row = r + 8 * hg;
      myP[row * 64 + ln]      = to_bf16_rz(p0);
      myP[row * 64 + 16 + ln] = to_bf16_rz(p1);
      myP[row * 64 + 32 + ln] = to_bf16_rz(p2);
      myP[row * 64 + 48 + ln] = to_bf16_rz(p3);
    }
#pragma unroll
    for (int f = 0; f < 4; ++f)
#pragma unroll
      for (int r = 0; r < 8; ++r) o[f][r] *= corr[r];

    asm volatile("s_wait_dscnt 0" ::: "memory");  // P C-layout -> A-layout transpose
    v16bf pa0 = load_a_frag(myP + ln * 64, hg * 16, hg * 16 + 8);
    v16bf pa1 = load_a_frag(myP + ln * 64, 32 + hg * 16, 40 + hg * 16);

    // O += P(16x64) . V_tile(64x64); V transposed so each lane load is contiguous
#pragma unroll
    for (int f = 0; f < 4; ++f) {
      const __bf16* vrow = Vp + (size_t)(16 * f + ln) * NN + j0;
      v16bf vb0 = *(const v16bf*)(vrow + hg * 16);
      o[f] = wmma_bf16(pa0, vb0, o[f]);
      v16bf vb1 = *(const v16bf*)(vrow + 32 + hg * 16);
      o[f] = wmma_bf16(pa1, vb1, o[f]);
    }
  }

  const int b = bh >> 3, h = bh & 7;
#pragma unroll
  for (int f = 0; f < 4; ++f)
#pragma unroll
    for (int r = 0; r < 8; ++r) {
      const int pos = q0 + r + 8 * hg;
      float val = o[f][r] / lst[r];
      Ob[((size_t)(b * NN + pos)) * CC + h * DD + 16 * f + ln] = to_bf16(val);
    }
}

// ------------------------------------------------- output projection GEMM (f32 out)
__global__ __launch_bounds__(256) void proj_kernel(const __bf16* __restrict__ Ab,
                                                   const __bf16* __restrict__ Wb,
                                                   const float* __restrict__ bias,
                                                   float* __restrict__ out) {
  const int wave = threadIdx.x >> 5, lane = threadIdx.x & 31;
  const int hg = lane >> 4, ln = lane & 15;
  const int wid = blockIdx.x * 8 + wave;        // 4096 wave-tiles
  const int m0 = (wid & 511) * 16;
  const int n0 = (wid >> 9) * 64;

  v8f acc[4];
#pragma unroll
  for (int f = 0; f < 4; ++f)
#pragma unroll
    for (int r = 0; r < 8; ++r) acc[f][r] = 0.0f;

  const __bf16* arow = Ab + (size_t)(m0 + ln) * CC;
  for (int kb = 0; kb < CC; kb += 32) {
    v16bf a = load_a_frag(arow, kb + hg * 16, kb + hg * 16 + 8);
#pragma unroll
    for (int f = 0; f < 4; ++f) {
      int col = n0 + 16 * f + ln;
      v16bf b = *(const v16bf*)(Wb + (size_t)col * CC + kb + hg * 16);
      acc[f] = wmma_bf16(a, b, acc[f]);
    }
  }
#pragma unroll
  for (int f = 0; f < 4; ++f) {
    const int col = n0 + 16 * f + ln;
    const float bs = bias[col];
#pragma unroll
    for (int r = 0; r < 8; ++r) {
      const int row = m0 + r + 8 * hg;
      out[(size_t)row * CC + col] = acc[f][r] + bs;
    }
  }
}

extern "C" void kernel_launch(void* const* d_in, const int* in_sizes, int n_in,
                              void* d_out, int out_size, void* d_ws, size_t ws_size,
                              hipStream_t stream) {
  (void)in_sizes; (void)n_in; (void)out_size; (void)ws_size;
  const float* x     = (const float*)d_in[0];
  const float* Wqkv  = (const float*)d_in[1];
  const float* bqkv  = (const float*)d_in[2];
  const float* Wproj = (const float*)d_in[3];
  const float* bproj = (const float*)d_in[4];
  float* out = (float*)d_out;

  char* ws = (char*)d_ws;
  __bf16* xb     = (__bf16*)(ws);                         //  8,388,608 B
  __bf16* wqkvb  = (__bf16*)(ws +  8388608);              //  1,572,864 B
  __bf16* wprojb = (__bf16*)(ws +  9961472);              //    524,288 B
  __bf16* Qb     = (__bf16*)(ws + 10485760);              //  8,388,608 B
  __bf16* Kb     = (__bf16*)(ws + 18874368);              //  8,388,608 B
  __bf16* Vt     = (__bf16*)(ws + 27262976);              //  8,388,608 B
  __bf16* Ob     = (__bf16*)(ws + 35651584);              //  8,388,608 B (end 44,040,192)

  cvt_bf16_kernel<<<2048, 256, 0, stream>>>(x,     xb,     BB * NN * CC);
  cvt_bf16_kernel<<<768,  256, 0, stream>>>(Wqkv,  wqkvb,  3 * CC * CC);
  cvt_bf16_kernel<<<256,  256, 0, stream>>>(Wproj, wprojb, CC * CC);

  qkv_rope_kernel<<<1536, 256, 0, stream>>>(xb, wqkvb, bqkv, Qb, Kb, Vt);
  attn_kernel<<<512, 256, 0, stream>>>(Qb, Kb, Vt, Ob);
  proj_kernel<<<512, 256, 0, stream>>>(Ob, wprojb, bproj, out);
}
